// GeometryPreservingContrastiveLoss_67233418051964
// MI455X (gfx1250) — matrix operand order repaired and checked by reading, not compiled, not run on hardware
//
#include <hip/hip_runtime.h>
#include <math.h>

// ---------------------------------------------------------------------------
// GeometryPreservingContrastiveLoss for MI455X (gfx1250), wave32 + WMMA bf16
// + Tensor Data Mover (tensor_load_to_lds, 6-arg clang-23 builtin) staging.
// B=512 images, K=100 boxes, D=1024, P=128, TEMP=0.07, IoU thresh 0.3.
// ---------------------------------------------------------------------------

typedef __bf16        v16bf __attribute__((ext_vector_type(16)));
typedef __bf16        v8bf  __attribute__((ext_vector_type(8)));
typedef float         v8f   __attribute__((ext_vector_type(8)));
typedef unsigned int  u32x4 __attribute__((ext_vector_type(4)));
typedef int           i32x8 __attribute__((ext_vector_type(8)));
typedef int           i32x4 __attribute__((ext_vector_type(4)));

#define MROWS 51200   // B*K
#define DDIM  1024
#define PDIM  128
#define BIMG  512
#define KBOX  100
#define KPAD  112     // 7 tiles of 16

__device__ __forceinline__ v8f wmma_bf16(v16bf a, v16bf b, v8f c) {
  return __builtin_amdgcn_wmma_f32_16x16x32_bf16(
      false, a, false, b, (short)0, c, false, false);
}

// A operand (16x32, MxK): lane holds row M=lane%16; K = lh*8+{0..7}, 16+lh*8+{0..7}
__device__ __forceinline__ v16bf ldA(const __bf16* rowBase, int lh) {
  const __bf16* p0 = rowBase + lh * 8;
  const __bf16* p1 = rowBase + 16 + lh * 8;
  v16bf r;
#pragma unroll
  for (int i = 0; i < 8; ++i) { r[i] = p0[i]; r[i + 8] = p1[i]; }
  return r;
}

// B operand (32x16, KxN): lane holds col N=lane%16; K = lh*8+{0..7}, 16+lh*8+{0..7}
__device__ __forceinline__ v16bf ldB(const __bf16* colBase, int lh) {
  const __bf16* p0 = colBase + lh * 8;
  const __bf16* p1 = colBase + 16 + lh * 8;
  v16bf r;
#pragma unroll
  for (int i = 0; i < 8; ++i) { r[i] = p0[i]; r[i + 8] = p1[i]; }
  return r;
}

// pack two floats into one dword of bf16 {hi<<16 | lo} (RNE via HW cvt)
__device__ __forceinline__ unsigned pack_bf2(float lo, float hi) {
  unsigned short lu = __builtin_bit_cast(unsigned short, (__bf16)lo);
  unsigned short hu = __builtin_bit_cast(unsigned short, (__bf16)hi);
  return ((unsigned)hu << 16) | (unsigned)lu;
}

// ---------------------------------------------------------------------------
// Tensor Data Mover: 2D bf16 tile load global->LDS with LDS padding.
// D# per CDNA5 ISA §8.3-8.6 (group0: count/lds/global/type, group1: dims).
// tensor dims == tile dims (tiles always fully in-bounds here).
// clang-23 toolchain: 6-arg builtin (g0, g1, g2, g3, g4, cpol).
// ---------------------------------------------------------------------------
__device__ __forceinline__ void tdm_load_2d_bf16(
    unsigned lds_off, const void* gptr,
    unsigned tile_d0, unsigned tile_d1, unsigned stride0_elems,
    unsigned pad_interval_code, unsigned pad_amount_code) {
  const unsigned long long ga = (unsigned long long)(uintptr_t)gptr;
  u32x4 g0;
  g0[0] = 1u;                                            // count=1 (valid user D#)
  g0[1] = lds_off;                                       // lds_addr (bytes)
  g0[2] = (unsigned)(ga & 0xFFFFFFFFu);                  // global_addr[31:0]
  g0[3] = (unsigned)((ga >> 32) & 0x01FFFFFFu) | (2u << 30);  // addr[56:32]|type=2
  i32x8 g1;
  unsigned w0 = (1u << 16)                               // data_size = 2 bytes
              | (1u << 20)                               // pad_enable
              | (pad_interval_code << 22)
              | (pad_amount_code << 25);
  g1[0] = (int)w0;
  g1[1] = (int)((tile_d0 & 0xFFFFu) << 16);              // tensor_dim0[15:0]
  g1[2] = (int)(((tile_d0 >> 16) & 0xFFFFu) | ((tile_d1 & 0xFFFFu) << 16)); // td0 hi | td1 lo
  g1[3] = (int)(((tile_d1 >> 16) & 0xFFFFu) | ((tile_d0 & 0xFFFFu) << 16)); // td1 hi | tile_dim0
  g1[4] = (int)(tile_d1 & 0xFFFFu);                      // tile_dim1 (tile_dim2=0)
  g1[5] = (int)stride0_elems;                            // tensor_dim0_stride[31:0]
  g1[6] = 0;                                             // stride0 hi | stride1 lo
  g1[7] = 0;                                             // stride1 hi
  const i32x4 gz4 = {0, 0, 0, 0};
  const i32x8 gz8 = {0, 0, 0, 0, 0, 0, 0, 0};
  __builtin_amdgcn_tensor_load_to_lds(g0, g1, gz4, gz4, gz8, 0);
}

// ---------------------------------------------------------------------------
// Kernel 1: H = relu(X @ W1 + b1), f32 in -> bf16 WMMA (f32 accum) -> bf16 H.
// Block tile 128(M)x64(N); 8 waves; each wave 32x32 (2x2 WMMA tiles); K step 32.
// ---------------------------------------------------------------------------
__global__ __launch_bounds__(256) void gemm1_relu_kernel(
    const float* __restrict__ X, const float* __restrict__ W1,
    const float* __restrict__ b1, __bf16* __restrict__ H) {
  __shared__ alignas(16) __bf16 sA[128][40];  // [m][k]
  __shared__ alignas(16) __bf16 sB[64][40];   // [n][k] (W1 tile transposed)

  const int t  = threadIdx.x;
  const int m0 = blockIdx.y * 128;
  const int n0 = blockIdx.x * 64;
  const int lane = t & 31, w = t >> 5;
  const int lm = lane & 15, lh = lane >> 4;
  const int wr = w >> 1, wc = w & 1;

  v8f c[2][2];
#pragma unroll
  for (int i = 0; i < 2; ++i)
#pragma unroll
    for (int j = 0; j < 2; ++j)
#pragma unroll
      for (int r = 0; r < 8; ++r) c[i][j][r] = 0.0f;

  const int arow = t >> 1, acol = (t & 1) * 16;  // A: 16 f32 -> 16 bf16 per thread
  const int bkp = t >> 4, bn0 = (t & 15) * 4;    // B: k-pair (2 rows) x 4 n per thread

  for (int k0 = 0; k0 < DDIM; k0 += 32) {
    // stage A: convert 16 floats -> packed b128 stores
    const float* gx = X + (size_t)(m0 + arow) * DDIM + k0 + acol;
    v8bf t0, t1;
#pragma unroll
    for (int i = 0; i < 8; ++i) { t0[i] = (__bf16)gx[i]; t1[i] = (__bf16)gx[8 + i]; }
    *(u32x4*)&sA[arow][acol]     = __builtin_bit_cast(u32x4, t0);
    *(u32x4*)&sA[arow][acol + 8] = __builtin_bit_cast(u32x4, t1);
    // stage B transposed: pack k-pairs -> b32 stores
    const float* gw0 = W1 + (size_t)(k0 + 2 * bkp) * DDIM + n0 + bn0;
    const float* gw1 = gw0 + DDIM;
#pragma unroll
    for (int i = 0; i < 4; ++i)
      *(unsigned*)&sB[bn0 + i][2 * bkp] = pack_bf2(gw0[i], gw1[i]);
    __syncthreads();

    v16bf bmat[2];
#pragma unroll
    for (int nt = 0; nt < 2; ++nt)
      bmat[nt] = ldB(&sB[wc * 32 + nt * 16 + lm][0], lh);
#pragma unroll
    for (int mt = 0; mt < 2; ++mt) {
      v16bf a = ldA(&sA[wr * 32 + mt * 16 + lm][0], lh);
#pragma unroll
      for (int nt = 0; nt < 2; ++nt)
        c[mt][nt] = wmma_bf16(a, bmat[nt], c[mt][nt]);
    }
    __syncthreads();
  }

#pragma unroll
  for (int mt = 0; mt < 2; ++mt)
#pragma unroll
    for (int nt = 0; nt < 2; ++nt) {
      const int col  = n0 + wc * 32 + nt * 16 + lm;
      const float bv = b1[col];
#pragma unroll
      for (int r = 0; r < 8; ++r) {
        const int row = m0 + wr * 32 + mt * 16 + r + 8 * lh;
        float v = c[mt][nt][r] + bv;
        v = v > 0.0f ? v : 0.0f;
        H[(size_t)row * DDIM + col] = (__bf16)v;
      }
    }
}

// ---------------------------------------------------------------------------
// Kernel 2: Z = normalize(H @ W2 + b2). A-tile (bf16 H) staged by the TDM with
// LDS padding, double-buffered so the DMA overlaps the WMMA loop.
// Block tile 128(M)x128(N=P); 8 waves; each wave 32x64 (2x4 tiles).
// ---------------------------------------------------------------------------
__global__ __launch_bounds__(256) void gemm2_norm_kernel(
    const __bf16* __restrict__ H, const float* __restrict__ W2,
    const float* __restrict__ b2, __bf16* __restrict__ Z) {
  __shared__ alignas(16) __bf16 sA[2][128][40];  // TDM-filled, stride 40 via pad
  __shared__ alignas(16) __bf16 sB[128][40];     // W2 tile transposed [n][k]
  __shared__ float sRsq[2][128];

  const int t  = threadIdx.x;
  const int m0 = blockIdx.x * 128;
  const int lane = t & 31, w = t >> 5;
  const int lm = lane & 15, lh = lane >> 4;
  const int wr = w >> 1, wc = w & 1;

  v8f c[2][4];
#pragma unroll
  for (int i = 0; i < 2; ++i)
#pragma unroll
    for (int j = 0; j < 4; ++j)
#pragma unroll
      for (int r = 0; r < 8; ++r) c[i][j][r] = 0.0f;

  const int bkp = t >> 4, bn0 = (t & 15) * 8;  // W2: k-pair x 8 n per thread

  // prime buffer 0: tile_dim0=32 bf16 = 16 dwords -> pad_interval code 3 (8x8B),
  // pad 4 dwords -> code 3  => LDS row stride 20 dwords = 40 bf16.
  if (w == 0)
    tdm_load_2d_bf16((unsigned)(uintptr_t)&sA[0][0][0],
                     H + (size_t)m0 * DDIM, 32, 128, DDIM, 3, 3);

  int buf = 0;
  for (int k0 = 0; k0 < DDIM; k0 += 32) {
    // stage B transposed (f32 -> packed bf16 pairs)
    const float* gw0 = W2 + (size_t)(k0 + 2 * bkp) * PDIM + bn0;
    const float* gw1 = gw0 + PDIM;
#pragma unroll
    for (int i = 0; i < 8; ++i)
      *(unsigned*)&sB[bn0 + i][2 * bkp] = pack_bf2(gw0[i], gw1[i]);
    if (w == 0) __builtin_amdgcn_s_wait_tensorcnt(0);  // A tile 'buf' complete
    __syncthreads();
    if (k0 + 32 < DDIM && w == 0)                      // overlap next DMA
      tdm_load_2d_bf16((unsigned)(uintptr_t)&sA[buf ^ 1][0][0],
                       H + (size_t)m0 * DDIM + (k0 + 32), 32, 128, DDIM, 3, 3);

    v16bf bmat[4];
#pragma unroll
    for (int nt = 0; nt < 4; ++nt)
      bmat[nt] = ldB(&sB[wc * 64 + nt * 16 + lm][0], lh);
#pragma unroll
    for (int mt = 0; mt < 2; ++mt) {
      v16bf a = ldA(&sA[buf][wr * 32 + mt * 16 + lm][0], lh);
#pragma unroll
      for (int nt = 0; nt < 4; ++nt)
        c[mt][nt] = wmma_bf16(a, bmat[nt], c[mt][nt]);
    }
    __syncthreads();
    buf ^= 1;
  }

  // bias add, then per-row sum of squares (rows split across 2 wave-columns)
  float bias[4];
#pragma unroll
  for (int nt = 0; nt < 4; ++nt) bias[nt] = b2[wc * 64 + nt * 16 + lm];
#pragma unroll
  for (int mt = 0; mt < 2; ++mt)
#pragma unroll
    for (int nt = 0; nt < 4; ++nt)
#pragma unroll
      for (int r = 0; r < 8; ++r) c[mt][nt][r] += bias[nt];

#pragma unroll
  for (int mt = 0; mt < 2; ++mt)
#pragma unroll
    for (int r = 0; r < 8; ++r) {
      float s = 0.0f;
#pragma unroll
      for (int nt = 0; nt < 4; ++nt) s += c[mt][nt][r] * c[mt][nt][r];
      s += __shfl_xor(s, 1, 32);
      s += __shfl_xor(s, 2, 32);
      s += __shfl_xor(s, 4, 32);
      s += __shfl_xor(s, 8, 32);
      if (lm == 0) sRsq[wc][wr * 32 + mt * 16 + r + 8 * lh] = s;
    }
  __syncthreads();

#pragma unroll
  for (int mt = 0; mt < 2; ++mt)
#pragma unroll
    for (int nt = 0; nt < 4; ++nt) {
      const int col = wc * 64 + nt * 16 + lm;
#pragma unroll
      for (int r = 0; r < 8; ++r) {
        const int row = wr * 32 + mt * 16 + r + 8 * lh;
        const float rs    = sRsq[0][row] + sRsq[1][row];
        const float scale = 1.0f / fmaxf(sqrtf(rs), 1e-12f);
        Z[(size_t)(m0 + row) * PDIM + col] = (__bf16)(c[mt][nt][r] * scale);
      }
    }
}

// ---------------------------------------------------------------------------
// Kernel 3: one block per image. z-tile staged by TDM (100x128 bf16, LDS
// stride 136 via pad). sim = z z^T / TEMP via WMMA; IoU + online LSE per row.
// ---------------------------------------------------------------------------
__global__ __launch_bounds__(256) void image_loss_kernel(
    const __bf16* __restrict__ Z, const float* __restrict__ boxes,
    float* __restrict__ perImg) {
  __shared__ alignas(16) __bf16 sZ[KPAD][136];
  __shared__ float sS[KPAD][116];
  __shared__ float sBox[KBOX][4];
  __shared__ float sArea[KBOX];
  __shared__ float sRed[256];

  const int t   = threadIdx.x;
  const int img = blockIdx.x;
  const int lane = t & 31, w = t >> 5;
  const int lm = lane & 15, lh = lane >> 4;

  // TDM: tile_dim0=128 bf16 = 64 dwords -> pad_interval code 5 (32x8B),
  // pad 4 dwords -> code 3  => LDS row stride 68 dwords = 136 bf16.
  if (w == 0)
    tdm_load_2d_bf16((unsigned)(uintptr_t)&sZ[0][0],
                     Z + (size_t)img * KBOX * PDIM, PDIM, KBOX, PDIM, 5, 3);

  // zero the pad rows (100..111)
  for (int idx = t; idx < 12 * 136; idx += 256)
    sZ[KBOX + idx / 136][idx % 136] = (__bf16)0.0f;
  // boxes + areas
  for (int idx = t; idx < KBOX; idx += 256) {
    const float4 bx = *(const float4*)(boxes + (size_t)(img * KBOX + idx) * 4);
    sBox[idx][0] = bx.x; sBox[idx][1] = bx.y;
    sBox[idx][2] = bx.z; sBox[idx][3] = bx.w;
    sArea[idx] = (bx.z - bx.x) * (bx.w - bx.y);
  }
  if (w == 0) __builtin_amdgcn_s_wait_tensorcnt(0);
  __syncthreads();

  // 7x7 = 49 sim tiles of 16x16, round-robin across 8 waves
  for (int tile = w; tile < 49; tile += 8) {
    const int ti = tile / 7, tj = tile % 7;
    v8f c;
#pragma unroll
    for (int r = 0; r < 8; ++r) c[r] = 0.0f;
#pragma unroll
    for (int ks = 0; ks < 4; ++ks) {
      v16bf a = ldA(&sZ[ti * 16 + lm][ks * 32], lh);
      v16bf b = ldB(&sZ[tj * 16 + lm][ks * 32], lh);
      c = wmma_bf16(a, b, c);
    }
#pragma unroll
    for (int r = 0; r < 8; ++r)
      sS[ti * 16 + r + 8 * lh][tj * 16 + lm] = c[r];
  }
  __syncthreads();

  float loss = 0.0f;
  if (t < KBOX) {
    const int i = t;
    const float x1 = sBox[i][0], y1 = sBox[i][1];
    const float x2 = sBox[i][2], y2 = sBox[i][3];
    const float ai = sArea[i];
    const float invT = 1.0f / 0.07f;
    float m = -INFINITY, s = 0.0f, possum = 0.0f, wsum = 0.0f;
    int npos = 0;
    for (int j = 0; j < KBOX; ++j) {
      if (j == i) continue;
      const float sim = sS[i][j] * invT;
      if (sim > m) { s = s * __expf(m - sim) + 1.0f; m = sim; }
      else         { s += __expf(sim - m); }
      const float lx = fmaxf(x1, sBox[j][0]);
      const float ly = fmaxf(y1, sBox[j][1]);
      const float rx = fminf(x2, sBox[j][2]);
      const float ry = fminf(y2, sBox[j][3]);
      const float iw = fmaxf(rx - lx, 0.0f);
      const float ih = fmaxf(ry - ly, 0.0f);
      const float inter = iw * ih;
      const float iou = inter / (ai + sArea[j] - inter);
      wsum += iou;
      if (iou > 0.3f) { ++npos; possum += sim; }
    }
    const bool valid = (wsum >= 1e-6f) && (npos > 0);
    loss = valid ? (-(possum / (float)npos) + (m + __logf(s))) : 0.0f;
  }
  sRed[t] = loss;
  for (int off = 128; off > 0; off >>= 1) {
    __syncthreads();
    if (t < off) sRed[t] += sRed[t + off];
  }
  __syncthreads();
  if (t == 0) perImg[img] = sRed[0] * (1.0f / (float)KBOX);
}

// ---------------------------------------------------------------------------
// Kernel 4: deterministic mean of 512 per-image losses.
// ---------------------------------------------------------------------------
__global__ __launch_bounds__(256) void reduce_mean_kernel(
    const float* __restrict__ perImg, float* __restrict__ out) {
  __shared__ float sred[256];
  const int t = threadIdx.x;
  sred[t] = perImg[t] + perImg[t + 256];
  for (int off = 128; off > 0; off >>= 1) {
    __syncthreads();
    if (t < off) sred[t] += sred[t + off];
  }
  __syncthreads();
  if (t == 0) out[0] = sred[0] * (1.0f / (float)BIMG);
}

// ---------------------------------------------------------------------------
extern "C" void kernel_launch(void* const* d_in, const int* in_sizes, int n_in,
                              void* d_out, int out_size, void* d_ws, size_t ws_size,
                              hipStream_t stream) {
  (void)in_sizes; (void)n_in; (void)out_size; (void)ws_size;
  const float* X     = (const float*)d_in[0];  // box_features [51200,1024]
  const float* boxes = (const float*)d_in[1];  // [51200,4]
  const float* W1    = (const float*)d_in[2];  // [1024,1024]
  const float* b1    = (const float*)d_in[3];  // [1024]
  const float* W2    = (const float*)d_in[4];  // [1024,128]
  const float* b2    = (const float*)d_in[5];  // [128]
  // d_in[6] = image_ids: contiguous K per image -> grouping is a reshape, unused.

  const size_t hBytes = (size_t)MROWS * DDIM * sizeof(__bf16);  // 104.9 MB
  const size_t zBytes = (size_t)MROWS * PDIM * sizeof(__bf16);  //  13.1 MB
  __bf16* H      = (__bf16*)d_ws;
  __bf16* Zn     = (__bf16*)((char*)d_ws + hBytes);
  float*  perImg = (float*)((char*)d_ws + hBytes + zBytes);

  dim3 g1(DDIM / 64, MROWS / 128);  // 16 x 400
  gemm1_relu_kernel<<<g1, 256, 0, stream>>>(X, W1, b1, H);
  gemm2_norm_kernel<<<MROWS / 128, 256, 0, stream>>>(H, W2, b2, Zn);
  image_loss_kernel<<<BIMG, 256, 0, stream>>>(Zn, boxes, perImg);
  reduce_mean_kernel<<<1, 256, 0, stream>>>(perImg, (float*)d_out);
}
